// VSAETopK_49770081026175
// MI455X (gfx1250) — compile-verified, matrix operand order checked by reading, not run on hardware
//
#include <hip/hip_runtime.h>

typedef unsigned short u16;
typedef __attribute__((ext_vector_type(16))) __bf16 v16bf;
typedef __attribute__((ext_vector_type(8)))  float  v8f;

constexpr int ACT   = 768;
constexpr int DICT  = 24576;
constexpr int BATCH = 8192;
constexpr int TOPK  = 64;

// ---- GEMM tiling ----
constexpr int BM = 128;
constexpr int BN = 256;
constexpr int BK = 32;
constexpr int NST = ACT / BK;        // 24 K-stages
constexpr int AP  = BK + 8;          // LDS pitch (u16): 40 -> 80B rows, conflict-free, 16B aligned
// LDS layout (u16 units):
//   A buffers: 2 x [128][40]  -> offsets 0, 5120
//   B buffers: 2 x [256][40]  -> offsets 10240, 20480
//   epilogue stage (reuses A+part of B after K loop): [64][264] = 16896 u16
constexpr int LDS_U16 = 30720;       // 61440 bytes
constexpr int AOFF = 0;
constexpr int ABUF = 5120;
constexpr int BOFF = 10240;
constexpr int BBUF = 10240;
constexpr int SP   = BN + 8;         // stage pitch 264

// ---------------------------------------------------------------- prep: f32 -> bf16 (RNE)
__device__ __forceinline__ u16 f32_to_bf16_rne(float f) {
  unsigned u = __float_as_uint(f);
  u += 0x7FFFu + ((u >> 16) & 1u);
  return (u16)(u >> 16);
}

__global__ void cvt_f32_bf16(const float* __restrict__ in, u16* __restrict__ out, int n) {
  int i = blockIdx.x * blockDim.x + threadIdx.x;
  if (i < n) out[i] = f32_to_bf16_rne(in[i]);
}

// ---------------------------------------------------------------- prep: W_dec [ACT][DICT] -> W_dec_T [DICT][ACT]
__global__ __launch_bounds__(256) void transpose_wdec(const float* __restrict__ in,
                                                      float* __restrict__ out) {
  __shared__ float tile[32][33];
  int lx = threadIdx.x & 31, ly = threadIdx.x >> 5;
  int d0 = blockIdx.x * 32;
  int a0 = blockIdx.y * 32;
#pragma unroll
  for (int i = 0; i < 4; ++i)
    tile[ly + i * 8][lx] = in[(size_t)(a0 + ly + i * 8) * DICT + d0 + lx];
  __syncthreads();
#pragma unroll
  for (int i = 0; i < 4; ++i)
    out[(size_t)(d0 + ly + i * 8) * ACT + a0 + lx] = tile[lx][ly + i * 8];
}

// ---------------------------------------------------------------- async global->LDS helpers (CDNA5, ASYNCcnt)
__device__ __forceinline__ void async_g2l_b128(unsigned lds_byte_off, const void* g) {
  asm volatile("global_load_async_to_lds_b128 %0, %1, off"
               :: "v"(lds_byte_off), "v"(g) : "memory");
}
__device__ __forceinline__ void wait_async_le6() {
  asm volatile("s_wait_asynccnt 0x6" ::: "memory");
}
__device__ __forceinline__ void wait_async_0() {
  asm volatile("s_wait_asynccnt 0x0" ::: "memory");
}

// ---------------------------------------------------------------- encode GEMM: mu16 = bf16(relu(x @ W_enc^T + b_enc))
union Frag { v16bf v; uint4 q[2]; };

__global__ __launch_bounds__(256) void sae_encode_gemm(const u16* __restrict__ X,
                                                       const u16* __restrict__ W,
                                                       const float* __restrict__ b_enc,
                                                       u16* __restrict__ mu) {
  __shared__ __align__(16) u16 lds[LDS_U16];

  const int tid  = threadIdx.x;
  const int lane = tid & 31;
  const int wid  = tid >> 5;
  const int wn   = wid & 3;            // 4 waves along N (64 cols each)
  const int wm   = wid >> 2;           // 2 waves along M (64 rows each)

  const int mblk = blockIdx.y * BM;
  const int nblk = blockIdx.x * BN;

  const unsigned lds_base = (unsigned)(uintptr_t)&lds[0];

  // per-thread async copy assignments (16B chunks)
  // A tile 128x32: 512 chunks -> 2/thread ; B tile 256x32: 1024 chunks -> 4/thread
  auto issue = [&](int kt, int b) {
#pragma unroll
    for (int c = 0; c < 2; ++c) {
      int ch = tid * 2 + c;
      int r = ch >> 2, kc = (ch & 3) * 8;
      unsigned loff = lds_base + (unsigned)(AOFF + b * ABUF + r * AP + kc) * 2u;
      async_g2l_b128(loff, X + (size_t)(mblk + r) * ACT + kt + kc);
    }
#pragma unroll
    for (int c = 0; c < 4; ++c) {
      int ch = tid * 4 + c;
      int r = ch >> 2, kc = (ch & 3) * 8;
      unsigned loff = lds_base + (unsigned)(BOFF + b * BBUF + r * AP + kc) * 2u;
      async_g2l_b128(loff, W + (size_t)(nblk + r) * ACT + kt + kc);
    }
  };

  const int mr   = lane & 15;
  const int krow = (lane >> 4) * 8;    // 0 or 8

  v8f acc[4][4];
#pragma unroll
  for (int mi = 0; mi < 4; ++mi)
#pragma unroll
    for (int ni = 0; ni < 4; ++ni)
      acc[mi][ni] = (v8f)0.0f;

  issue(0, 0);
  for (int s = 0; s < NST; ++s) {
    if (s + 1 < NST) { issue((s + 1) * BK, (s + 1) & 1); wait_async_le6(); }
    else             { wait_async_0(); }
    __syncthreads();

    const int b = s & 1;
    Frag fa[4];
#pragma unroll
    for (int mi = 0; mi < 4; ++mi) {
      const u16* p = &lds[AOFF + b * ABUF + (wm * 64 + mi * 16 + mr) * AP];
      fa[mi].q[0] = *reinterpret_cast<const uint4*>(p + krow);
      fa[mi].q[1] = *reinterpret_cast<const uint4*>(p + krow + 16);
    }
#pragma unroll
    for (int ni = 0; ni < 4; ++ni) {
      Frag fb;
      const u16* p = &lds[BOFF + b * BBUF + (wn * 64 + ni * 16 + mr) * AP];
      fb.q[0] = *reinterpret_cast<const uint4*>(p + krow);
      fb.q[1] = *reinterpret_cast<const uint4*>(p + krow + 16);
#pragma unroll
      for (int mi = 0; mi < 4; ++mi)
        acc[mi][ni] = __builtin_amdgcn_wmma_f32_16x16x32_bf16(
            false, fa[mi].v, false, fb.v, (short)0, acc[mi][ni], false, false);
    }
    __syncthreads();
  }

  // ---- epilogue: +b_enc, relu, bf16, LDS-staged coalesced stores (two 64-row halves) ----
  float be[4];
#pragma unroll
  for (int ni = 0; ni < 4; ++ni)
    be[ni] = b_enc[nblk + wn * 64 + ni * 16 + (lane & 15)];

  for (int half = 0; half < 2; ++half) {
    if (wm == half) {
#pragma unroll
      for (int mi = 0; mi < 4; ++mi)
#pragma unroll
        for (int ni = 0; ni < 4; ++ni) {
          const int cc = wn * 64 + ni * 16 + (lane & 15);
#pragma unroll
          for (int j = 0; j < 8; ++j) {
            const int r = mi * 16 + j + ((lane >> 4) * 8);   // 0..63 within half
            float v = acc[mi][ni][j] + be[ni];
            lds[r * SP + cc] = f32_to_bf16_rne(v > 0.0f ? v : 0.0f);
          }
        }
    }
    __syncthreads();
    // readback: 64 rows x 256 cols -> 2048 16B chunks, 8 per thread, coalesced b128 stores
#pragma unroll
    for (int c = 0; c < 8; ++c) {
      int ch = tid * 8 + c;
      int r  = ch >> 5;
      int cc = (ch & 31) * 8;
      uint4 q = *reinterpret_cast<const uint4*>(&lds[r * SP + cc]);
      *reinterpret_cast<uint4*>(mu + (size_t)(mblk + half * 64 + r) * DICT + nblk + cc) = q;
    }
    __syncthreads();
  }
}

// ---------------------------------------------------------------- top-64 select + sparse decode, one block per row
__global__ __launch_bounds__(256) void topk_decode(const u16* __restrict__ mu,
                                                   const float* __restrict__ WdecT,
                                                   const float* __restrict__ b_dec,
                                                   float* __restrict__ out) {
  __shared__ __align__(16) u16 codes[DICT];   // bf16 bits, sign=0 -> monotone u16 keys
  __shared__ unsigned hist[256];
  __shared__ unsigned cntg[256], cnte[256];
  __shared__ float svals[TOPK];
  __shared__ int   sidx[TOPK];
  __shared__ unsigned s_B, s_base, s_P, s_G, s_rem;

  const int row = blockIdx.x;
  const int tid = threadIdx.x;
  const u16* r = mu + (size_t)row * DICT;

  // pass 1: vectorized copy to LDS + histogram of high byte
  hist[tid] = 0;
  __syncthreads();
  for (int i = tid * 8; i < DICT; i += 256 * 8) {
    uint4 q = *reinterpret_cast<const uint4*>(r + i);
    *reinterpret_cast<uint4*>(&codes[i]) = q;
    unsigned w[4] = {q.x, q.y, q.z, q.w};
#pragma unroll
    for (int k = 0; k < 4; ++k) {
      atomicAdd(&hist[(w[k] & 0xFFFFu) >> 8], 1u);
      atomicAdd(&hist[w[k] >> 24], 1u);
    }
  }
  __syncthreads();
  if (tid == 0) {
    unsigned cum = 0; int B = 0;
    for (int b = 255; b >= 0; --b) {
      if (cum + hist[b] >= TOPK) { B = b; break; }
      cum += hist[b];
    }
    s_B = (unsigned)B; s_base = cum;
  }
  __syncthreads();
  const unsigned B = s_B, base = s_base;

  // pass 2: refine low byte within bucket B
  hist[tid] = 0;
  __syncthreads();
  for (int i = tid; i < DICT; i += 256) {
    u16 c = codes[i];
    if ((unsigned)(c >> 8) == B) atomicAdd(&hist[c & 0xFF], 1u);
  }
  __syncthreads();
  if (tid == 0) {
    unsigned cum = base; int B2 = 0;
    for (int b = 255; b >= 0; --b) {
      if (cum + hist[b] >= TOPK) { B2 = b; break; }
      cum += hist[b];
    }
    s_P = (B << 8) | (unsigned)B2;
    s_G = cum;
    s_rem = TOPK - cum;
  }
  __syncthreads();
  const unsigned P = s_P, G = s_G, rem = s_rem;

  // pass 3: deterministic selection (contiguous chunks + exclusive scan, index-ordered ties)
  constexpr int CH = DICT / 256;   // 96
  const int i0 = tid * CH;
  unsigned cg = 0, ce = 0;
  for (int i = i0; i < i0 + CH; ++i) {
    u16 c = codes[i];
    cg += (c > P);
    ce += (c == P);
  }
  cntg[tid] = cg; cnte[tid] = ce;
  __syncthreads();
  if (tid == 0) {
    unsigned sg = 0, se = 0;
    for (int t = 0; t < 256; ++t) {
      unsigned a = cntg[t], b = cnte[t];
      cntg[t] = sg; cnte[t] = se;
      sg += a; se += b;
    }
  }
  __syncthreads();
  unsigned pg = cntg[tid], pe = cnte[tid];
  for (int i = i0; i < i0 + CH; ++i) {
    u16 c = codes[i];
    if (c > P) {
      sidx[pg++] = i;
    } else if (c == P) {
      unsigned q = pe++;
      if (q < rem) sidx[G + q] = i;
    }
  }
  __syncthreads();
  if (tid < TOPK) svals[tid] = __uint_as_float((unsigned)codes[sidx[tid]] << 16);
  __syncthreads();

  // decode: x_hat[row,a] = b_dec[a] + sum_j svals[j] * WdecT[sidx[j]][a]
  for (int a = tid; a < ACT; a += 256) {
    float acc = b_dec[a];
#pragma unroll 8
    for (int j = 0; j < TOPK; ++j)
      acc = fmaf(svals[j], WdecT[(size_t)sidx[j] * ACT + a], acc);
    out[(size_t)row * ACT + a] = acc;
  }
}

// ---------------------------------------------------------------- host launch
extern "C" void kernel_launch(void* const* d_in, const int* in_sizes, int n_in,
                              void* d_out, int out_size, void* d_ws, size_t ws_size,
                              hipStream_t stream) {
  const float* x     = (const float*)d_in[0];
  const float* W_enc = (const float*)d_in[1];
  const float* b_enc = (const float*)d_in[2];
  const float* W_dec = (const float*)d_in[3];
  const float* b_dec = (const float*)d_in[4];
  float* out = (float*)d_out;

  char* ws = (char*)d_ws;
  size_t off = 0;
  auto alloc = [&](size_t bytes) { char* p = ws + off; off = (off + bytes + 255) & ~(size_t)255; return p; };
  u16*   x_bf  = (u16*)  alloc((size_t)BATCH * ACT * sizeof(u16));
  u16*   w_bf  = (u16*)  alloc((size_t)DICT  * ACT * sizeof(u16));
  float* wdecT = (float*)alloc((size_t)DICT  * ACT * sizeof(float));
  u16*   mu16  = (u16*)  alloc((size_t)BATCH * DICT * sizeof(u16));

  {
    int n = BATCH * ACT;
    cvt_f32_bf16<<<(n + 255) / 256, 256, 0, stream>>>(x, x_bf, n);
  }
  {
    int n = DICT * ACT;
    cvt_f32_bf16<<<(n + 255) / 256, 256, 0, stream>>>(W_enc, w_bf, n);
  }
  transpose_wdec<<<dim3(DICT / 32, ACT / 32), 256, 0, stream>>>(W_dec, wdecT);

  sae_encode_gemm<<<dim3(DICT / BN, BATCH / BM), 256, 0, stream>>>(x_bf, w_bf, b_enc, mu16);

  topk_decode<<<BATCH, 256, 0, stream>>>(mu16, wdecT, b_dec, out);
}
